// MultiheadDiffAttention_75342316306983
// MI455X (gfx1250) — matrix operand order repaired and checked by reading, not compiled, not run on hardware
//
#include <hip/hip_runtime.h>
#include <math.h>

// ---------------------------------------------------------------------------
// MI455X (gfx1250) Multihead Differential Attention
//   B=4, T=1024, E=1024, H=16 head-pairs, half-head dim D=32, v dim 2D=64.
// ~52 GFLOP vs ~100MB HBM traffic -> compute-bound -> bf16 WMMA
// (v_wmma_f32_16x16x32_bf16, f32 accum) for every matmul; fp32 softmax.
// Whole working set (~50MB) fits in 192MB L2, so GEMMs stream from L2 with
// register double-buffering; attention stages K/V tiles to LDS with
// global_load_async_to_lds_b128 (ASYNCcnt) when the toolchain exposes it.
// ---------------------------------------------------------------------------

typedef __attribute__((ext_vector_type(16))) __bf16 v16bf;
typedef __attribute__((ext_vector_type(8)))  float  v8f;

#define Bn 4
#define Tn 1024
#define En 1024
#define Hn 16
#define Dn 32
#define LAMBDA_INIT 0.7836057666f           // 0.8 - 0.6*exp(-0.3*12)
#define ONE_MINUS_LAMBDA_INIT 0.2163942334f
#define QK_SCALE 0.17677669529663687f       // 32^-0.5

#if defined(__has_builtin)
#if __has_builtin(__builtin_amdgcn_global_load_async_to_lds_b128)
#define HAVE_ASYNC_LDS 1
#endif
#endif

#ifdef HAVE_ASYNC_LDS
// Builtin signature (from hipcc diagnostic): pointee is int32x4, with
// global (__device__/AS1) source and LDS (AS3) destination + imm offset/cpol.
typedef int v4i_vs __attribute__((__vector_size__(16)));
typedef __attribute__((address_space(1))) v4i_vs GV4;
typedef __attribute__((address_space(3))) v4i_vs LV4;
static __device__ __forceinline__ void async_copy16(const void* g, void* l) {
  // 16B global -> LDS, no VGPR staging, tracked by ASYNCcnt
  __builtin_amdgcn_global_load_async_to_lds_b128((GV4*)g, (LV4*)l, 0, 0);
}
#if __has_builtin(__builtin_amdgcn_s_wait_asynccnt)
#define WAIT_ASYNC() __builtin_amdgcn_s_wait_asynccnt(0)
#else
#define WAIT_ASYNC() asm volatile("s_wait_asynccnt 0" ::: "memory")
#endif
#else
#define WAIT_ASYNC()
#endif

// wave-private LDS RAW fence: DS ops are in-order per wave
#if defined(__has_builtin)
#if __has_builtin(__builtin_amdgcn_s_wait_dscnt)
#define WAIT_DS() __builtin_amdgcn_s_wait_dscnt(0)
#else
#define WAIT_DS() asm volatile("s_wait_dscnt 0" ::: "memory")
#endif
#else
#define WAIT_DS() asm volatile("s_wait_dscnt 0" ::: "memory")
#endif

// 16-lane group reductions (rows of a WMMA f32 C-fragment live in lanes 0-15
// or 16-31; xor masks 1,2,4,8 never cross the group boundary).
static __device__ __forceinline__ float redmax16(float x) {
  x = fmaxf(x, __shfl_xor(x, 1, 32));
  x = fmaxf(x, __shfl_xor(x, 2, 32));
  x = fmaxf(x, __shfl_xor(x, 4, 32));
  x = fmaxf(x, __shfl_xor(x, 8, 32));
  return x;
}
static __device__ __forceinline__ float redsum16(float x) {
  x += __shfl_xor(x, 1, 32);
  x += __shfl_xor(x, 2, 32);
  x += __shfl_xor(x, 4, 32);
  x += __shfl_xor(x, 8, 32);
  return x;
}

// A-fragment loader (16-bit A 16x32, contiguous k order, not the ISA table order):
//   lane = (m | half<<4); element e -> k = half*16 + e
// Arow must already include the half*16 offset. Compiles to 2x b128.
static __device__ __forceinline__ v16bf load_afrag(const __bf16* __restrict__ Arow,
                                                   int kk) {
  v16bf a;
#pragma unroll
  for (int e = 0; e < 8; ++e)  a[e] = Arow[kk + e];
#pragma unroll
  for (int e = 8; e < 16; ++e) a[e] = Arow[kk + e];
  return a;
}

// ---------------------------------------------------------------------------
// fp32 -> bf16 cast
// ---------------------------------------------------------------------------
__global__ void cast_bf16_kernel(const float* __restrict__ src,
                                 __bf16* __restrict__ dst, int n) {
  int i = blockIdx.x * blockDim.x + threadIdx.x;
  if (i < n) dst[i] = (__bf16)src[i];
}

// ---------------------------------------------------------------------------
// C[M,N] = alpha * A[M,K] @ W[N,K]^T   (torch Linear convention)
// One wave per 32x64 C tile, 8 WMMAs per 32-wide K step, software-pipelined
// with two register fragment buffers so global loads overlap the WMMAs.
// ---------------------------------------------------------------------------
__global__ __launch_bounds__(32)
void gemm_wmma_kernel(const __bf16* __restrict__ A,
                      const __bf16* __restrict__ W,
                      float alpha,
                      __bf16* __restrict__ Cb,   // bf16 out (or null)
                      float*  __restrict__ Cf,   // f32 out  (or null)
                      int M, int N, int Kd) {
  const int lane = threadIdx.x & 31;
  const int mn   = lane & 15;
  const int half = lane >> 4;
  const int n0 = blockIdx.x * 64;
  const int m0 = blockIdx.y * 32;

  v8f acc[2][4] = {};

  const __bf16* Ar[2];
  Ar[0] = A + (size_t)(m0 + mn) * Kd + half * 16;
  Ar[1] = Ar[0] + (size_t)16 * Kd;
  const __bf16* Wr[4];
#pragma unroll
  for (int c = 0; c < 4; ++c)
    Wr[c] = W + (size_t)(n0 + c * 16 + mn) * Kd + half * 16;

  v16bf a0[2], w0[4], a1[2], w1[4];
#pragma unroll
  for (int p = 0; p < 2; ++p) a0[p] = load_afrag(Ar[p], 0);
#pragma unroll
  for (int c = 0; c < 4; ++c) w0[c] = *(const v16bf*)(Wr[c]);

  for (int kk = 0; kk < Kd; kk += 64) {
    const int k1 = kk + 32;                       // Kd is a multiple of 64
#pragma unroll
    for (int p = 0; p < 2; ++p) a1[p] = load_afrag(Ar[p], k1);
#pragma unroll
    for (int c = 0; c < 4; ++c) w1[c] = *(const v16bf*)(Wr[c] + k1);
#pragma unroll
    for (int p = 0; p < 2; ++p)
#pragma unroll
      for (int c = 0; c < 4; ++c)
        acc[p][c] = __builtin_amdgcn_wmma_f32_16x16x32_bf16(
            false, a0[p], false, w0[c], (short)0, acc[p][c], false, false);

    const int k2 = (kk + 64 < Kd) ? kk + 64 : 0;  // stay in-bounds on last iter
#pragma unroll
    for (int p = 0; p < 2; ++p) a0[p] = load_afrag(Ar[p], k2);
#pragma unroll
    for (int c = 0; c < 4; ++c) w0[c] = *(const v16bf*)(Wr[c] + k2);
#pragma unroll
    for (int p = 0; p < 2; ++p)
#pragma unroll
      for (int c = 0; c < 4; ++c)
        acc[p][c] = __builtin_amdgcn_wmma_f32_16x16x32_bf16(
            false, a1[p], false, w1[c], (short)0, acc[p][c], false, false);
  }

#pragma unroll
  for (int p = 0; p < 2; ++p)
#pragma unroll
    for (int c = 0; c < 4; ++c)
#pragma unroll
      for (int r = 0; r < 8; ++r) {
        int row = m0 + p * 16 + r + half * 8;   // f32 C: M split across halves
        int col = n0 + c * 16 + mn;
        float v = acc[p][c][r] * alpha;
        if (Cf) Cf[(size_t)row * N + col] = v;
        else    Cb[(size_t)row * N + col] = (__bf16)v;
      }
}

// ---------------------------------------------------------------------------
// Fused flash-style differential attention.
// Grid: Bn*Hn*(Tn/32) blocks, 128 threads (4 waves).
//   wave w: score map j = w&1 (half-head 2h+j), q-subtile qsub = w>>1.
// K/V tiles double-buffered in LDS; staged with async-to-LDS b128 where
// available (issue tile i+1, compute tile i, s_wait_asynccnt + barrier).
// Per 32-key tile each wave: S = Q x K^T (2 WMMAs), online softmax on the
// f32 C fragment, P->bf16 LDS round-trip (wave-private, s_wait_dscnt),
// O += P(16x32) x V(32x16) (4 WMMAs).
// Epilogue: o1 - lam*o2, per-row RMSNorm over 64 ch, *g, *(1-lambda_init).
// ---------------------------------------------------------------------------
__global__ __launch_bounds__(128)
void diffattn_wmma_kernel(const __bf16* __restrict__ Q,   // [B*T, E] pre-scaled
                          const __bf16* __restrict__ K,   // [B*T, E]
                          const __bf16* __restrict__ V,   // [B*T, E]
                          __bf16* __restrict__ O,         // [B*T, E]
                          const float* __restrict__ lq1, const float* __restrict__ lk1,
                          const float* __restrict__ lq2, const float* __restrict__ lk2,
                          const float* __restrict__ g) {
  __shared__ __align__(64) __bf16 Kt[2][2][32][32];  // [buf][map][key][d]  8KB
  __shared__ __align__(64) __bf16 Vt[2][32][64];     // [buf][key][vcol]    8KB
  __shared__ __align__(64) __bf16 Pt[4][16][32];     // per-wave P tile     4KB
  __shared__ __align__(64) float  Ot[2][16][64];     // map-1 publish       8KB
  __shared__ float lamsh;

  const int tid  = threadIdx.x;
  const int lane = tid & 31;
  const int wv   = tid >> 5;
  const int jmap = wv & 1;
  const int qsub = wv >> 1;
  const int mn   = lane & 15;
  const int half = lane >> 4;

  const int bid = blockIdx.x;
  const int qt  = bid & 31;            // Tn/32 q tiles
  const int h   = (bid >> 5) & (Hn - 1);
  const int b   = bid >> 9;

  if (tid == 0) {
    float s1 = 0.f, s2 = 0.f;
    for (int i = 0; i < Dn; ++i) { s1 += lq1[i] * lk1[i]; s2 += lq2[i] * lk2[i]; }
    lamsh = __expf(s1) - __expf(s2) + LAMBDA_INIT;
  }

  const size_t rowbase = (size_t)b * Tn;
  const int q0 = qt * 32 + qsub * 16;

  // Stage one 32-key tile (K for both maps + V) into LDS buffer `bufs`.
  auto stage_tiles = [&](int k0s, int bufs) {
#ifdef HAVE_ASYNC_LDS
    // 256 x 16B chunks each for K and V; 128 threads -> 2 chunks per array.
    for (int id = tid; id < 256; id += 128) {
      int jj = id >> 7, key = (id >> 2) & 31, d0 = (id & 3) * 8;
      async_copy16(K + (rowbase + k0s + key) * En + (2 * h + jj) * Dn + d0,
                   &Kt[bufs][jj][key][d0]);
    }
    for (int id = tid; id < 256; id += 128) {
      int key = id >> 3, c0 = (id & 7) * 8;
      async_copy16(V + (rowbase + k0s + key) * En + h * 64 + c0,
                   &Vt[bufs][key][c0]);
    }
#else
    for (int idx = tid; idx < 2 * 32 * 32; idx += 128) {
      int jj = idx >> 10, key = (idx >> 5) & 31, d = idx & 31;
      Kt[bufs][jj][key][d] =
          K[(rowbase + k0s + key) * En + (2 * h + jj) * Dn + d];
    }
    for (int idx = tid; idx < 32 * 64; idx += 128) {
      int key = idx >> 6, c = idx & 63;
      Vt[bufs][key][c] = V[(rowbase + k0s + key) * En + h * 64 + c];
    }
#endif
  };

  // Q A-fragment (16 q-rows x 32 dims = full half-head: K == D == 32)
  v16bf qa = load_afrag(
      Q + (rowbase + q0 + mn) * En + (2 * h + jmap) * Dn + half * 16, 0);

  v8f acc[4] = {};
  float Mrow[8], Lrow[8];
#pragma unroll
  for (int r = 0; r < 8; ++r) { Mrow[r] = -1e30f; Lrow[r] = 0.f; }

  stage_tiles(0, 0);
  WAIT_ASYNC();
  __syncthreads();

  int buf = 0;
  for (int k0 = 0; k0 < Tn; k0 += 32) {
    if (k0 + 32 < Tn) stage_tiles(k0 + 32, buf ^ 1);   // overlap with compute

#pragma unroll
    for (int sub = 0; sub < 2; ++sub) {
      // B-fragment of K^T: lane holds K[key = sub*16+mn][half*16 .. +16)
      v16bf kb = *(const v16bf*)&Kt[buf][jmap][sub * 16 + mn][half * 16];
      v8f zero = {};
      v8f s = __builtin_amdgcn_wmma_f32_16x16x32_bf16(
          false, qa, false, kb, (short)0, zero, false, false);

#pragma unroll
      for (int r = 0; r < 8; ++r) {
        float x    = s[r];
        float tmax = redmax16(x);
        float newM = fmaxf(Mrow[r], tmax);
        float scal = __expf(Mrow[r] - newM);
        float p    = __expf(x - newM);
        float psum = redsum16(p);
        Lrow[r] = Lrow[r] * scal + psum;
        Mrow[r] = newM;
#pragma unroll
        for (int c = 0; c < 4; ++c) acc[c][r] *= scal;
        // C-layout row r+half*8, col = key-in-tile
        Pt[wv][r + half * 8][sub * 16 + mn] = (__bf16)p;
      }
    }
    WAIT_DS();   // Pt is wave-private; DS in-order per wave -> no block barrier

    // A-fragment of P (16x32 bf16)
    v16bf pa;
#pragma unroll
    for (int e = 0; e < 8; ++e)  pa[e] = Pt[wv][mn][half * 16 + e];
#pragma unroll
    for (int e = 8; e < 16; ++e) pa[e] = Pt[wv][mn][half * 16 + e];

#pragma unroll
    for (int c = 0; c < 4; ++c) {
      // B-fragment of V (32 keys x 16 vcols), row-major tile -> strided gather
      v16bf vb;
#pragma unroll
      for (int e = 0; e < 16; ++e) vb[e] = Vt[buf][half * 16 + e][c * 16 + mn];
      acc[c] = __builtin_amdgcn_wmma_f32_16x16x32_bf16(
          false, pa, false, vb, (short)0, acc[c], false, false);
    }

    WAIT_ASYNC();        // next tile's async copies landed
    __syncthreads();     // all waves done reading buf / writing buf^1
    buf ^= 1;
  }

  // Normalize by softmax denominators
#pragma unroll
  for (int r = 0; r < 8; ++r) {
    float inv = 1.0f / Lrow[r];
#pragma unroll
    for (int c = 0; c < 4; ++c) acc[c][r] *= inv;
  }

  __syncthreads();
  if (jmap == 1) {  // publish second softmax map's output
#pragma unroll
    for (int c = 0; c < 4; ++c)
#pragma unroll
      for (int r = 0; r < 8; ++r)
        Ot[qsub][r + half * 8][c * 16 + mn] = acc[c][r];
  }
  __syncthreads();

  if (jmap == 0) {
    const float lam = lamsh;
    float val[4][8];
#pragma unroll
    for (int c = 0; c < 4; ++c)
#pragma unroll
      for (int r = 0; r < 8; ++r)
        val[c][r] = acc[c][r] - lam * Ot[qsub][r + half * 8][c * 16 + mn];

    float gc[4];
#pragma unroll
    for (int c = 0; c < 4; ++c) gc[c] = g[c * 16 + mn];

#pragma unroll
    for (int r = 0; r < 8; ++r) {
      float ss = val[0][r] * val[0][r] + val[1][r] * val[1][r] +
                 val[2][r] * val[2][r] + val[3][r] * val[3][r];
      ss = redsum16(ss);                       // full 64-channel row sum
      float rn = rsqrtf(ss * (1.0f / 64.0f) + 1e-5f) * ONE_MINUS_LAMBDA_INIT;
      size_t orow = (rowbase + q0 + r + half * 8) * (size_t)En + h * 64;
#pragma unroll
      for (int c = 0; c < 4; ++c)
        O[orow + c * 16 + mn] = (__bf16)(val[c][r] * rn * gc[c]);
    }
  }
}

// ---------------------------------------------------------------------------
// Launch: cast -> Q/K/V projections (WMMA) -> fused diff-attn (WMMA)
//         -> output projection (WMMA, f32 into d_out).
// Workspace: 48 MiB of bf16 staging.
// ---------------------------------------------------------------------------
extern "C" void kernel_launch(void* const* d_in, const int* in_sizes, int n_in,
                              void* d_out, int out_size, void* d_ws, size_t ws_size,
                              hipStream_t stream) {
  (void)in_sizes; (void)n_in; (void)out_size; (void)ws_size;

  const float* x   = (const float*)d_in[0];
  const float* Wq  = (const float*)d_in[1];
  const float* Wk  = (const float*)d_in[2];
  const float* Wv  = (const float*)d_in[3];
  const float* Wo  = (const float*)d_in[4];
  const float* lq1 = (const float*)d_in[5];
  const float* lk1 = (const float*)d_in[6];
  const float* lq2 = (const float*)d_in[7];
  const float* lk2 = (const float*)d_in[8];
  const float* g   = (const float*)d_in[9];
  float* out = (float*)d_out;

  const size_t MT = (size_t)Bn * Tn;         // 4096 rows
  char* wsb = (char*)d_ws;
  size_t off = 0;
  auto take = [&](size_t elems) { __bf16* p = (__bf16*)(wsb + off); off += elems * 2; return p; };
  __bf16* xb  = take(MT * En);
  __bf16* wqb = take((size_t)En * En);
  __bf16* wkb = take((size_t)En * En);
  __bf16* wvb = take((size_t)En * En);
  __bf16* wob = take((size_t)En * En);
  __bf16* Qb  = take(MT * En);
  __bf16* Kb  = take(MT * En);
  __bf16* Vb  = take(MT * En);
  __bf16* Ab  = take(MT * En);

  // Casts
  {
    int n = (int)(MT * En);
    cast_bf16_kernel<<<(n + 255) / 256, 256, 0, stream>>>(x, xb, n);
    int nw = En * En;
    dim3 gw((nw + 255) / 256);
    cast_bf16_kernel<<<gw, 256, 0, stream>>>(Wq, wqb, nw);
    cast_bf16_kernel<<<gw, 256, 0, stream>>>(Wk, wkb, nw);
    cast_bf16_kernel<<<gw, 256, 0, stream>>>(Wv, wvb, nw);
    cast_bf16_kernel<<<gw, 256, 0, stream>>>(Wo, wob, nw);
  }

  // Projections: one wave per 32x64 tile
  dim3 ggrid(En / 64, (unsigned)(MT / 32));
  gemm_wmma_kernel<<<ggrid, 32, 0, stream>>>(xb, wqb, QK_SCALE, Qb, nullptr,
                                             (int)MT, En, En);
  gemm_wmma_kernel<<<ggrid, 32, 0, stream>>>(xb, wkb, 1.0f, Kb, nullptr,
                                             (int)MT, En, En);
  gemm_wmma_kernel<<<ggrid, 32, 0, stream>>>(xb, wvb, 1.0f, Vb, nullptr,
                                             (int)MT, En, En);

  // Fused differential attention: B*H*(T/32) = 2048 blocks, 4 waves each
  diffattn_wmma_kernel<<<Bn * Hn * (Tn / 32), 128, 0, stream>>>(
      Qb, Kb, Vb, Ab, lq1, lk1, lq2, lk2, g);

  // Output projection, fp32 result
  gemm_wmma_kernel<<<ggrid, 32, 0, stream>>>(Ab, wob, 1.0f, nullptr, out,
                                             (int)MT, En, En);
}
